// CausalSelfAttention_77661598646700
// MI455X (gfx1250) — matrix operand order repaired and checked by reading, not compiled, not run on hardware
//
#include <hip/hip_runtime.h>
#include <hip/hip_bf16.h>

#define D_MODEL   1024
#define NUM_HEADS 16
#define HEAD_DIM  64
#define BATCH     2
#define SEQ       2048
#define M_ROWS    (BATCH * SEQ)   // 4096

typedef __bf16 bf16_t;
typedef bf16_t v16bf  __attribute__((ext_vector_type(16)));
typedef bf16_t bf16x8 __attribute__((ext_vector_type(8)));
typedef float  v8f    __attribute__((ext_vector_type(8)));
typedef unsigned int u32x4 __attribute__((ext_vector_type(4)));
typedef int          i32x4 __attribute__((ext_vector_type(4)));
typedef int          i32x8 __attribute__((ext_vector_type(8)));

union Frag { v16bf v; bf16x8 h[2]; };

// --- Tensor Data Mover availability / arity detection -----------------------
#if defined(__has_builtin)
#  if __has_builtin(__builtin_amdgcn_tensor_load_to_lds)
#    define USE_TDM 1
#    if __has_include(<hip/amd_detail/amd_gfx1250_TDM.h>)
#      define TDM_6ARG 1   // amdgpu-toolchain (clang-23): 6-arg builtin
#    endif
#  endif
#endif

#if defined(USE_TDM)
// Issue a TDM load of a 2D bf16 tile [tile_rows x 64] from a row-major
// [tensor_rows x 1024] matrix into LDS at byte offset lds_byte.
// D# layout per CDNA5 ISA ch.8 (group0: count/lds/global/type, group1: dims).
__device__ __forceinline__ void tdm_load_tile(const void* gbase, unsigned lds_byte,
                                              int tile_rows, int tensor_rows) {
    unsigned long long ga = (unsigned long long)gbase;
    u32x4 g0;
    g0[0] = 1u;                                               // count=1, user desc
    g0[1] = lds_byte;                                         // lds_addr (bytes)
    g0[2] = (unsigned)ga;                                     // global_addr[31:0]
    g0[3] = (unsigned)((ga >> 32) & 0x01FFFFFFull) | (2u << 30); // [56:32] | type=2
    i32x8 g1;
    g1[0] = (1 << 16);                                        // data_size=1 (2 bytes)
    g1[1] = (int)((D_MODEL & 0xFFFF) << 16);                  // tensor_dim0 lo16
    g1[2] = (int)((tensor_rows & 0xFFFF) << 16);              // dim0 hi16=0 | dim1 lo16
    g1[3] = (64 << 16);                                       // dim1 hi16=0 | tile_dim0=64
    g1[4] = tile_rows;                                        // tile_dim1 | tile_dim2=0
    g1[5] = D_MODEL;                                          // tensor_dim0_stride lo32
    g1[6] = 0;                                                // stride hi | dim1_stride lo
    g1[7] = 0;
    i32x4 z4 = {0, 0, 0, 0};
#if defined(TDM_6ARG)
    i32x8 z8 = {0, 0, 0, 0, 0, 0, 0, 0};
    __builtin_amdgcn_tensor_load_to_lds(g0, g1, z4, z4, z8, 0);
#else
    __builtin_amdgcn_tensor_load_to_lds(g0, g1, z4, z4, 0);
#endif
}
#endif

// ---------------------------------------------------------------------------
// fp32 -> bf16 elementwise convert (x activations)
// ---------------------------------------------------------------------------
__global__ __launch_bounds__(256)
void cvt_bf16_kernel(const float* __restrict__ src, bf16_t* __restrict__ dst, int n) {
    int i = (blockIdx.x * 256 + threadIdx.x) * 4;
    if (i + 3 < n) {
        float4 f = *(const float4*)(src + i);
        dst[i + 0] = (bf16_t)f.x;
        dst[i + 1] = (bf16_t)f.y;
        dst[i + 2] = (bf16_t)f.z;
        dst[i + 3] = (bf16_t)f.w;
    }
}

// ---------------------------------------------------------------------------
// fp32 W[K,N] -> bf16 Wt[N,K]  (transpose so B-frags read contiguous K)
// ---------------------------------------------------------------------------
__global__ __launch_bounds__(256)
void tr_cvt_kernel(const float* __restrict__ src, bf16_t* __restrict__ dst) {
    int idx = blockIdx.x * 256 + threadIdx.x;     // 0 .. 1M-1
    int k = idx & (D_MODEL - 1);
    int n = idx >> 10;
    dst[idx] = (bf16_t)src[k * D_MODEL + n];      // dst[n*K + k]
}

// ---------------------------------------------------------------------------
// GEMM: C[M=4096, N=1024] = A[M,K=1024] * Bt[N,K]^T (+bias)*alpha
// Block 256 thr = 8 waves in 2(M) x 4(N) grid of 32x32 wave tiles.
// TDM path: A tile (64x64) and B tile (128x64) DMA'd into double-buffered LDS
// by wave 0 while all waves run 8 WMMAs per stage (TENSORcnt pipeline; one
// barrier per 64-deep K stage). Grid: (M/64, N/128) = (64, 8).
// ---------------------------------------------------------------------------
#define MODE_HEADS   0  // bf16 out -> [b,h,s,d]
#define MODE_HEADS_T 1  // bf16 out -> [b,h,d,s]  (V transposed)
#define MODE_OUT_F32 2  // fp32 row-major [M,N]

__global__ __launch_bounds__(256)
void gemm_wmma_kernel(const bf16_t* __restrict__ A,
                      const bf16_t* __restrict__ Bt,
                      const float*  __restrict__ bias,
                      void* __restrict__ dst,
                      int mode, float alpha) {
    const int K = D_MODEL, N = D_MODEL;
    const int lane = threadIdx.x & 31;
    const int wid  = threadIdx.x >> 5;
    const int g    = lane >> 4;
    const int ln   = lane & 15;
    const int wm   = wid & 1;
    const int wn   = wid >> 1;
    const int row0 = blockIdx.x * 64  + wm * 32;
    const int col0 = blockIdx.y * 128 + wn * 32;

    const v8f vzero = {0.f, 0.f, 0.f, 0.f, 0.f, 0.f, 0.f, 0.f};
    v8f acc[2][2];
    for (int i = 0; i < 2; i++)
        for (int j = 0; j < 2; j++) acc[i][j] = vzero;

#if defined(USE_TDM)
    // Single shared array so LDS byte offsets used in the TDM descriptors
    // line up with the ds_load pointers:
    //   A double buffer: bytes [0, 8K) and [8K, 16K)    (64 rows x 64 k)
    //   B double buffer: bytes [16K, 32K) and [32K, 48K) (128 rows x 64 k)
    __shared__ bf16_t smem[2 * 64 * 64 + 2 * 128 * 64];     // 48 KB
    bf16_t* Abuf = smem;
    bf16_t* Bbuf = smem + 2 * 64 * 64;

    const bf16_t* Ablk = A  + (size_t)(blockIdx.x * 64)  * K;   // 64 rows
    const bf16_t* Bblk = Bt + (size_t)(blockIdx.y * 128) * K;   // 128 rows
    const int T = K / 64;                                       // 16 stages

    if (wid == 0) {
        tdm_load_tile(Ablk, 0u, 64, M_ROWS);
        tdm_load_tile(Bblk, 16384u, 128, N);
        __builtin_amdgcn_s_wait_tensorcnt(0);
    }
    __syncthreads();

    for (int t = 0; t < T; t++) {
        const int p = t & 1;
        // The TDM engine wrote smem: escape the pointer into an asm with a
        // memory clobber so the compiler treats the LDS tiles as defined and
        // keeps (and re-issues each stage) the ds_load frag reads.
        asm volatile("" :: "v"(smem) : "memory");

        if (t + 1 < T && wid == 0) {   // DMA next tiles while computing this one
            tdm_load_tile(Ablk + (size_t)(t + 1) * 64, (unsigned)((1 - p) * 8192), 64, M_ROWS);
            tdm_load_tile(Bblk + (size_t)(t + 1) * 64, (unsigned)(16384 + (1 - p) * 16384), 128, N);
        }

        // two 32-deep WMMA sub-steps per 64-deep stage (8 WMMAs between syncs)
        for (int u = 0; u < 2; u++) {
            Frag a[2], b[2];
            for (int i = 0; i < 2; i++) {
                const bf16_t* ar = Abuf + p * (64 * 64) + (wm * 32 + i * 16 + ln) * 64 + u * 32;
                a[i].h[0] = *(const bf16x8*)(ar + 8 * g);
                a[i].h[1] = *(const bf16x8*)(ar + 8 * g + 16);
                const bf16_t* br = Bbuf + p * (128 * 64) + (wn * 32 + i * 16 + ln) * 64 + u * 32;
                b[i].h[0] = *(const bf16x8*)(br + 8 * g);
                b[i].h[1] = *(const bf16x8*)(br + 8 * g + 16);
            }
            for (int i = 0; i < 2; i++)
                for (int j = 0; j < 2; j++)
                    acc[i][j] = __builtin_amdgcn_wmma_f32_16x16x32_bf16(
                        false, a[i].v, false, b[j].v, (short)0, acc[i][j], false, false);
        }

        if (t + 1 < T) {
            if (wid == 0) __builtin_amdgcn_s_wait_tensorcnt(0);
            __syncthreads();
        }
    }
#else
    // Fallback: direct global-load register GEMM (proven path)
    const bf16_t* aptr[2];
    const bf16_t* bptr[2];
    aptr[0] = A  + (size_t)(row0 + ln)      * K;
    aptr[1] = A  + (size_t)(row0 + 16 + ln) * K;
    bptr[0] = Bt + (size_t)(col0 + ln)      * K;
    bptr[1] = Bt + (size_t)(col0 + 16 + ln) * K;
    for (int kk = 0; kk < K; kk += 32) {
        Frag a[2], b[2];
        for (int i = 0; i < 2; i++) {
            a[i].h[0] = *(const bf16x8*)(aptr[i] + kk + 8 * g);
            a[i].h[1] = *(const bf16x8*)(aptr[i] + kk + 8 * g + 16);
            b[i].h[0] = *(const bf16x8*)(bptr[i] + kk + 8 * g);
            b[i].h[1] = *(const bf16x8*)(bptr[i] + kk + 8 * g + 16);
        }
        for (int i = 0; i < 2; i++)
            for (int j = 0; j < 2; j++)
                acc[i][j] = __builtin_amdgcn_wmma_f32_16x16x32_bf16(
                    false, a[i].v, false, b[j].v, (short)0, acc[i][j], false, false);
    }
#endif

    // Epilogue: D element v of lane(g,ln) -> row M = v + 8g, col N = ln
    for (int i = 0; i < 2; i++) {
        for (int j = 0; j < 2; j++) {
            int col = col0 + j * 16 + ln;
            float bv = bias[col];
            for (int v = 0; v < 8; v++) {
                int row = row0 + i * 16 + v + 8 * g;
                float val = (acc[i][j][v] + bv) * alpha;
                if (mode == MODE_OUT_F32) {
                    ((float*)dst)[(size_t)row * N + col] = val;
                } else {
                    int b = row >> 11, s = row & (SEQ - 1);
                    int h = col >> 6,  d = col & 63;
                    size_t bh = (size_t)(b * NUM_HEADS + h);
                    if (mode == MODE_HEADS)
                        ((bf16_t*)dst)[(bh * SEQ + s) * HEAD_DIM + d] = (bf16_t)val;
                    else
                        ((bf16_t*)dst)[(bh * HEAD_DIM + d) * SEQ + s] = (bf16_t)val;
                }
            }
        }
    }
}

// ---------------------------------------------------------------------------
// Flash attention (causal). One wave = one 16-row Q tile, k-tiles of 32.
// Q [bh,S,64] pre-scaled by 1/8; K [bh,S,64]; V transposed [bh,64,S].
// Out bf16 [b,s, h*64+d] = [M_ROWS, D_MODEL].
// Block 256 (8 waves), grid (S/128, B*H) = (16, 32).
// ---------------------------------------------------------------------------
__global__ __launch_bounds__(256)
void flash_attn_kernel(const bf16_t* __restrict__ Q,
                       const bf16_t* __restrict__ Km,
                       const bf16_t* __restrict__ Vt,
                       bf16_t* __restrict__ Out) {
    __shared__ bf16_t Pl[8 * 16 * 32];   // per-wave 16x32 bf16 P tile
    const int lane = threadIdx.x & 31;
    const int wid  = threadIdx.x >> 5;
    const int g    = lane >> 4;
    const int ln   = lane & 15;
    const int bh   = blockIdx.y;
    const int b    = bh >> 4;
    const int h    = bh & 15;
    const int q0   = (blockIdx.x * 8 + wid) * 16;

    const bf16_t* Qb = Q  + (size_t)bh * SEQ * HEAD_DIM;
    const bf16_t* Kb = Km + (size_t)bh * SEQ * HEAD_DIM;
    const bf16_t* Vb = Vt + (size_t)bh * HEAD_DIM * SEQ;
    bf16_t* Pw = Pl + wid * 16 * 32;

    // Q A-frags for contraction d = 0..63 (two 32-wide k-steps), kept resident
    Frag aq[2];
    {
        const bf16_t* qr = Qb + (size_t)(q0 + ln) * HEAD_DIM;
        for (int t = 0; t < 2; t++) {
            aq[t].h[0] = *(const bf16x8*)(qr + t * 32 + 8 * g);
            aq[t].h[1] = *(const bf16x8*)(qr + t * 32 + 8 * g + 16);
        }
    }

    const v8f vzero = {0.f, 0.f, 0.f, 0.f, 0.f, 0.f, 0.f, 0.f};
    v8f O[4];
    for (int j = 0; j < 4; j++) O[j] = vzero;
    float rm[8], rl[8];
    for (int v = 0; v < 8; v++) { rm[v] = -1e30f; rl[v] = 0.f; }

    const int nk = (q0 >> 5) + 1;   // causal: only k-tiles with kb <= q0+15
    for (int kt = 0; kt < nk; kt++) {
        const int kb = kt * 32;
        if (kt + 1 < nk) {  // prefetch next K/V tiles -> global_prefetch_b8
            __builtin_prefetch(Kb + (size_t)(kb + 32 + ln) * HEAD_DIM, 0, 0);
            __builtin_prefetch(Vb + (size_t)ln * SEQ + kb + 32, 0, 0);
        }

        // scores S[16 x 32] = Q * K^T : two 16-col halves
        v8f s[2];
        for (int half = 0; half < 2; half++) {
            const int krow = kb + half * 16 + ln;
            const bf16_t* kr = Kb + (size_t)krow * HEAD_DIM;
            v8f sh = vzero;
            for (int t = 0; t < 2; t++) {
                Frag bk;
                bk.h[0] = *(const bf16x8*)(kr + t * 32 + 8 * g);
                bk.h[1] = *(const bf16x8*)(kr + t * 32 + 8 * g + 16);
                sh = __builtin_amdgcn_wmma_f32_16x16x32_bf16(
                    false, aq[t].v, false, bk.v, (short)0, sh, false, false);
            }
            s[half] = sh;
        }

        // causal mask + tile row-max (rows live in VGPR v across 16 lanes of group g)
        float tmax[8];
        for (int v = 0; v < 8; v++) {
            const int qi = q0 + v + 8 * g;
            float s0 = s[0][v], s1 = s[1][v];
            if (kb + ln > qi)      s0 = -1e30f;
            if (kb + 16 + ln > qi) s1 = -1e30f;
            s[0][v] = s0; s[1][v] = s1;
            tmax[v] = fmaxf(s0, s1);
        }
        for (int m = 1; m < 16; m <<= 1)
            for (int v = 0; v < 8; v++)
                tmax[v] = fmaxf(tmax[v], __shfl_xor(tmax[v], m, 32));

        // online softmax update; write P (bf16) to per-wave LDS tile
        float tsum[8];
        for (int v = 0; v < 8; v++) {
            float mn = fmaxf(rm[v], tmax[v]);
            float f  = __expf(rm[v] - mn);
            rm[v] = mn;
            float p0 = __expf(s[0][v] - mn);
            float p1 = __expf(s[1][v] - mn);
            rl[v] *= f;
            for (int j = 0; j < 4; j++) O[j][v] *= f;
            int r = v + 8 * g;
            Pw[r * 32 + ln]      = (bf16_t)p0;
            Pw[r * 32 + 16 + ln] = (bf16_t)p1;
            tsum[v] = p0 + p1;
        }
        for (int m = 1; m < 16; m <<= 1)
            for (int v = 0; v < 8; v++)
                tsum[v] += __shfl_xor(tsum[v], m, 32);
        for (int v = 0; v < 8; v++) rl[v] += tsum[v];

        asm volatile("s_wait_dscnt 0" ::: "memory");   // LDS stores -> loads

        // P as A-frag (transpose via LDS), then O += P * V
        Frag ap;
        ap.h[0] = *(const bf16x8*)(Pw + ln * 32 + 8 * g);
        ap.h[1] = *(const bf16x8*)(Pw + ln * 32 + 8 * g + 16);
        for (int j = 0; j < 4; j++) {
            const bf16_t* vp = Vb + (size_t)(j * 16 + ln) * SEQ + kb + 8 * g;
            Frag bv;
            bv.h[0] = *(const bf16x8*)(vp);
            bv.h[1] = *(const bf16x8*)(vp + 16);
            O[j] = __builtin_amdgcn_wmma_f32_16x16x32_bf16(
                false, ap.v, false, bv.v, (short)0, O[j], false, false);
        }
    }

    // normalize and store to [b, s, h*64+d] (bf16) for the output projection
    for (int v = 0; v < 8; v++) {
        float inv = 1.0f / rl[v];
        int r = q0 + v + 8 * g;
        size_t base = ((size_t)b * SEQ + r) * D_MODEL + h * HEAD_DIM;
        for (int j = 0; j < 4; j++)
            Out[base + j * 16 + ln] = (bf16_t)(O[j][v] * inv);
    }
}

// ---------------------------------------------------------------------------
extern "C" void kernel_launch(void* const* d_in, const int* in_sizes, int n_in,
                              void* d_out, int out_size, void* d_ws, size_t ws_size,
                              hipStream_t stream) {
    const float* x  = (const float*)d_in[0];
    const float* Wq = (const float*)d_in[1];
    const float* bq = (const float*)d_in[2];
    const float* Wk = (const float*)d_in[3];
    const float* bk = (const float*)d_in[4];
    const float* Wv = (const float*)d_in[5];
    const float* bv = (const float*)d_in[6];
    const float* Wo = (const float*)d_in[7];
    const float* bo = (const float*)d_in[8];

    char* ws = (char*)d_ws;
    const size_t act_bytes = (size_t)M_ROWS * D_MODEL * sizeof(bf16_t);   // 8 MB
    const size_t w_bytes   = (size_t)D_MODEL * D_MODEL * sizeof(bf16_t);  // 2 MB
    const size_t hd_bytes  = (size_t)BATCH * NUM_HEADS * SEQ * HEAD_DIM * sizeof(bf16_t); // 8 MB

    bf16_t* x_bf    = (bf16_t*)ws; ws += act_bytes;
    bf16_t* wq_t    = (bf16_t*)ws; ws += w_bytes;
    bf16_t* wk_t    = (bf16_t*)ws; ws += w_bytes;
    bf16_t* wv_t    = (bf16_t*)ws; ws += w_bytes;
    bf16_t* wo_t    = (bf16_t*)ws; ws += w_bytes;
    bf16_t* q_bf    = (bf16_t*)ws; ws += hd_bytes;
    bf16_t* k_bf    = (bf16_t*)ws; ws += hd_bytes;
    bf16_t* vt_bf   = (bf16_t*)ws; ws += hd_bytes;
    bf16_t* attn_bf = (bf16_t*)ws; ws += act_bytes;

    // 1. convert activations, transpose+convert weights
    const int n_x = M_ROWS * D_MODEL;                       // 4,194,304
    cvt_bf16_kernel<<<n_x / (256 * 4), 256, 0, stream>>>(x, x_bf, n_x);
    const int n_w = D_MODEL * D_MODEL;                      // 1,048,576
    tr_cvt_kernel<<<n_w / 256, 256, 0, stream>>>(Wq, wq_t);
    tr_cvt_kernel<<<n_w / 256, 256, 0, stream>>>(Wk, wk_t);
    tr_cvt_kernel<<<n_w / 256, 256, 0, stream>>>(Wv, wv_t);
    tr_cvt_kernel<<<n_w / 256, 256, 0, stream>>>(Wo, wo_t);

    // 2. Q/K/V projections (Q pre-scaled by 1/sqrt(64); V stored transposed)
    dim3 ggrid(M_ROWS / 64, D_MODEL / 128);
    gemm_wmma_kernel<<<ggrid, 256, 0, stream>>>(x_bf, wq_t, bq, q_bf,  MODE_HEADS,   0.125f);
    gemm_wmma_kernel<<<ggrid, 256, 0, stream>>>(x_bf, wk_t, bk, k_bf,  MODE_HEADS,   1.0f);
    gemm_wmma_kernel<<<ggrid, 256, 0, stream>>>(x_bf, wv_t, bv, vt_bf, MODE_HEADS_T, 1.0f);

    // 3. causal flash attention
    dim3 agrid(SEQ / 128, BATCH * NUM_HEADS);
    flash_attn_kernel<<<agrid, 256, 0, stream>>>(q_bf, k_bf, vt_bf, attn_bf);

    // 4. output projection -> fp32 d_out
    gemm_wmma_kernel<<<ggrid, 256, 0, stream>>>(attn_bf, wo_t, bo, d_out, MODE_OUT_F32, 1.0f);
}
